// ViTWithSoftHGNN_SeS_21869973471349
// MI455X (gfx1250) — matrix operand order repaired and checked by reading, not compiled, not run on hardware
//
#include <hip/hip_runtime.h>
#include <math.h>

// ---------------- problem constants ----------------
#define B_        32
#define C_        3
#define IMG_      224
#define P_        16
#define D_        768
#define DEPTH_    12
#define HEADS_    12
#define DH_       64
#define MLP_      3072
#define NCLS_     1000
#define E_DYN_    32
#define TOPK_     16
#define E_FIX_    16
#define EH_       8
#define E_TOT_    48
#define EHD_      96
#define GRID_     14
#define NPATCH_   196
#define SEQ_      197
#define SLD_      208        // padded leading dim for attention scores (16B-aligned rows)
#define PATCH_DIM_ 3840
#define LN_EPS_   1e-5f

typedef _Float16 v16h __attribute__((ext_vector_type(16)));
typedef _Float16 h8   __attribute__((ext_vector_type(8)));
typedef float    v8f  __attribute__((ext_vector_type(8)));
typedef float    f4   __attribute__((ext_vector_type(4)));

__device__ __forceinline__ float gelu_f(float x) {
    return 0.5f * x * (1.0f + erff(x * 0.70710678118654752f));
}
__device__ __forceinline__ int clampi(int v, int hi) { return v < hi ? v : hi; }

// =====================================================================
// WMMA GEMM: C = act(alpha * (opA(A) @ opB(B)) + bias) [+ resid]
// 256 threads = 8 waves. Block tile 64(M) x 64(N), K step 32.
// Each wave: one A fragment, two 16x16 accumulators (2 wmma / K-step).
// LDS layouts are "fragment order": each lane reads its 16-half fragment
// as two contiguous b128 loads (conflict-free with 40-half row stride).
//   lA[m][pk]  pk = ((k>>3)&1)*16 + (k>>4)*8 + (k&7)   (A operand layout)
//   lBt[n][k]  (B operand layout: lane k-half is contiguous)
// Edge handling is branchless: clamped addresses + value select.
// Next K-tile is prefetched (global_prefetch_b8) while WMMA runs.
// Two-level batch: z -> (z/batch2, z%batch2).
// =====================================================================
template<int TA, int TB, int ACT, int RESID>
__global__ __launch_bounds__(256) void wmma_gemm_kernel(
    const float* __restrict__ A, const float* __restrict__ Bm,
    const float* __restrict__ bias, const float* __restrict__ resid,
    float* __restrict__ C,
    int M, int N, int K, int lda, int ldb, int ldc,
    float alpha, int batch2,
    long long sA1, long long sA2, long long sB1, long long sB2,
    long long sC1, long long sC2)
{
    __shared__ alignas(16) _Float16 lA [64 * 40];
    __shared__ alignas(16) _Float16 lBt[64 * 40];

    const int z  = blockIdx.z;
    const int z1 = z / batch2, z2 = z % batch2;
    const float* Ab = A  + z1 * sA1 + z2 * sA2;
    const float* Bb = Bm + z1 * sB1 + z2 * sB2;
    const long long offC = z1 * sC1 + z2 * sC2;

    const int m0 = blockIdx.y * 64;
    const int n0 = blockIdx.x * 64;
    const int t    = threadIdx.x;
    const int lane = t & 31;
    const int wid  = t >> 5;
    const int wm   = wid & 3;      // M sub-tile 0..3
    const int wn   = wid >> 2;     // N half 0..1 (32 cols each)
    const int hs   = lane >> 4;    // lane half select
    const int lrow = lane & 15;

    const bool mfull = (m0 + 64 <= M);
    const bool nfull = (n0 + 64 <= N);

    v8f acc0 = {}, acc1 = {};

    for (int k0 = 0; k0 < K; k0 += 32) {
        const bool kfull = (k0 + 32 <= K);

        // ---- prefetch next K tile while this one is staged/computed ----
        if (k0 + 32 < K) {
            const int kn = k0 + 32;
            if (TA == 0)
                __builtin_prefetch(Ab + (long long)clampi(m0 + (t >> 2), M - 1) * lda
                                      + clampi(kn + (t & 3) * 8, K - 1), 0, 0);
            if (TB == 0)
                __builtin_prefetch(Bb + (long long)clampi(kn + (t >> 6) * 8, K - 1) * ldb
                                      + clampi(n0 + (t & 63), N - 1), 0, 0);
            else
                __builtin_prefetch(Bb + (long long)clampi(n0 + (t >> 2), N - 1) * ldb
                                      + clampi(kn + (t & 3) * 8, K - 1), 0, 0);
        }

        // ---------------- stage A tile (64 x 32) ----------------
        if (TA == 0) {
            const int r  = t >> 2;                       // 0..63 (m)
            const int c0 = (t & 3) * 8;                  // 0,8,16,24 (k)
            const int pb = ((c0 >> 3) & 1) * 16 + ((c0 >> 4) & 1) * 8;
            h8 hv;
            if (mfull && kfull) {
                const float* src = Ab + (long long)(m0 + r) * lda + (k0 + c0);
                const f4 x0 = *(const f4*)(src);
                const f4 x1 = *(const f4*)(src + 4);
#pragma unroll
                for (int j = 0; j < 4; ++j) { hv[j] = (_Float16)x0[j]; hv[j+4] = (_Float16)x1[j]; }
            } else {
                const int  m  = m0 + r;
                const bool mi = (m < M);
                const float* srow = Ab + (long long)clampi(m, M - 1) * lda;
#pragma unroll
                for (int j = 0; j < 8; ++j) {
                    const int k = k0 + c0 + j;
                    const float v = srow[clampi(k, K - 1)];
                    hv[j] = (_Float16)((mi && k < K) ? v : 0.0f);
                }
            }
            *(h8*)&lA[r * 40 + pb] = hv;
        } else { // TA == 1 : A stored K x M (minor GEMMs only)
            const int kk  = t >> 3;                      // 0..31 local k
            const int mc0 = (t & 7) * 8;                 // m offset
            const int pk  = ((kk >> 3) & 1) * 16 + ((kk >> 4) & 1) * 8 + (kk & 7);
            const int  k  = k0 + kk;
            const bool ki = (k < K);
            const float* srow = Ab + (long long)clampi(k, K - 1) * lda;
#pragma unroll
            for (int j = 0; j < 8; ++j) {
                const int m = m0 + mc0 + j;
                const float v = srow[clampi(m, M - 1)];
                lA[(mc0 + j) * 40 + pk] = (_Float16)((ki && m < M) ? v : 0.0f);
            }
        }

        // ---------------- stage B tile (32 x 64) -> lBt[n][k] ----------------
        if (TB == 0) { // B stored K x N: coalesce along n, accumulate along k
            const int nn = t & 63;                       // 0..63 (n)
            const int r0 = (t >> 6) * 8;                 // 0,8,16,24 (k)
            h8 hv;
            if (nfull && kfull) {
                const float* src = Bb + (long long)(k0 + r0) * ldb + (n0 + nn);
#pragma unroll
                for (int j = 0; j < 8; ++j) hv[j] = (_Float16)src[(long long)j * ldb];
            } else {
                const int  n  = n0 + nn;
                const bool ni = (n < N);
                const float* scol = Bb + clampi(n, N - 1);
#pragma unroll
                for (int j = 0; j < 8; ++j) {
                    const int k = k0 + r0 + j;
                    const float v = scol[(long long)clampi(k, K - 1) * ldb];
                    hv[j] = (_Float16)((ni && k < K) ? v : 0.0f);
                }
            }
            *(h8*)&lBt[nn * 40 + r0] = hv;
        } else {       // B stored N x K: contiguous along k
            const int nn  = t >> 2;                      // 0..63 (n)
            const int kc0 = (t & 3) * 8;                 // k offset
            h8 hv;
            if (nfull && kfull) {
                const float* src = Bb + (long long)(n0 + nn) * ldb + (k0 + kc0);
                const f4 x0 = *(const f4*)(src);
                const f4 x1 = *(const f4*)(src + 4);
#pragma unroll
                for (int j = 0; j < 4; ++j) { hv[j] = (_Float16)x0[j]; hv[j+4] = (_Float16)x1[j]; }
            } else {
                const int  n  = n0 + nn;
                const bool ni = (n < N);
                const float* srow = Bb + (long long)clampi(n, N - 1) * ldb;
#pragma unroll
                for (int j = 0; j < 8; ++j) {
                    const int k = k0 + kc0 + j;
                    const float v = srow[clampi(k, K - 1)];
                    hv[j] = (_Float16)((ni && k < K) ? v : 0.0f);
                }
            }
            *(h8*)&lBt[nn * 40 + kc0] = hv;
        }
        __syncthreads();

        // ---------------- fragments: two contiguous b128 loads each ----------------
        const _Float16* ap = &lA[(wm * 16 + lrow) * 40 + hs * 8];
        const h8 a_lo = *(const h8*)(ap);
        const h8 a_hi = *(const h8*)(ap + 16);
        const v16h af = __builtin_shufflevector(a_lo, a_hi,
            0,1,2,3,4,5,6,7,8,9,10,11,12,13,14,15);

        const _Float16* bp0 = &lBt[(wn * 32 + lrow) * 40 + hs * 16];
        const h8 b0_lo = *(const h8*)(bp0);
        const h8 b0_hi = *(const h8*)(bp0 + 8);
        const v16h bf0 = __builtin_shufflevector(b0_lo, b0_hi,
            0,1,2,3,4,5,6,7,8,9,10,11,12,13,14,15);

        const _Float16* bp1 = &lBt[(wn * 32 + 16 + lrow) * 40 + hs * 16];
        const h8 b1_lo = *(const h8*)(bp1);
        const h8 b1_hi = *(const h8*)(bp1 + 8);
        const v16h bf1 = __builtin_shufflevector(b1_lo, b1_hi,
            0,1,2,3,4,5,6,7,8,9,10,11,12,13,14,15);

        acc0 = __builtin_amdgcn_wmma_f32_16x16x32_f16(
            false, af, false, bf0, (short)0, acc0, false, false);
        acc1 = __builtin_amdgcn_wmma_f32_16x16x32_f16(
            false, af, false, bf1, (short)0, acc1, false, false);
        __syncthreads();
    }

    // ---------------- epilogue ----------------
    if (mfull && nfull) {
#pragma unroll
        for (int s = 0; s < 2; ++s) {
            const v8f& acc = s ? acc1 : acc0;
            const int gcol = n0 + wn * 32 + s * 16 + lrow;
            const float bv = bias ? bias[gcol] : 0.0f;
#pragma unroll
            for (int r = 0; r < 8; ++r) {
                const int grow = m0 + wm * 16 + hs * 8 + r;
                float v = acc[r] * alpha + bv;
                if (ACT == 1) v = gelu_f(v);
                const long long ci = offC + (long long)grow * ldc + gcol;
                if (RESID) v += resid[ci];
                C[ci] = v;
            }
        }
    } else {
#pragma unroll
        for (int s = 0; s < 2; ++s) {
            const v8f& acc = s ? acc1 : acc0;
            const int gcol = n0 + wn * 32 + s * 16 + lrow;
#pragma unroll
            for (int r = 0; r < 8; ++r) {
                const int grow = m0 + wm * 16 + hs * 8 + r;
                if (grow < M && gcol < N) {
                    float v = acc[r] * alpha;
                    if (bias) v += bias[gcol];
                    if (ACT == 1) v = gelu_f(v);
                    const long long ci = offC + (long long)grow * ldc + gcol;
                    if (RESID) v += resid[ci];
                    C[ci] = v;
                }
            }
        }
    }
}

static void launch_gemm(hipStream_t st, const float* A, const float* Bm,
                        const float* bias, const float* resid, float* C,
                        int M, int N, int K, int lda, int ldb, int ldc,
                        int transA, int transB, float alpha, int act,
                        int nbatch, int batch2,
                        long long sA1, long long sA2,
                        long long sB1, long long sB2,
                        long long sC1, long long sC2)
{
    dim3 g((N + 63) / 64, (M + 63) / 64, nbatch);
#define GEMM_ARGS A, Bm, bias, resid, C, M, N, K, lda, ldb, ldc, alpha, batch2, \
                  sA1, sA2, sB1, sB2, sC1, sC2
    if (transA == 0 && transB == 0) {
        if      (act == 0 && resid == nullptr) wmma_gemm_kernel<0,0,0,0><<<g,256,0,st>>>(GEMM_ARGS);
        else if (act == 1 && resid == nullptr) wmma_gemm_kernel<0,0,1,0><<<g,256,0,st>>>(GEMM_ARGS);
        else if (act == 0)                     wmma_gemm_kernel<0,0,0,1><<<g,256,0,st>>>(GEMM_ARGS);
        else                                   wmma_gemm_kernel<0,0,1,1><<<g,256,0,st>>>(GEMM_ARGS);
    } else if (transA == 0 && transB == 1) {
        wmma_gemm_kernel<0,1,0,0><<<g,256,0,st>>>(GEMM_ARGS);
    } else {
        wmma_gemm_kernel<1,0,0,0><<<g,256,0,st>>>(GEMM_ARGS);
    }
#undef GEMM_ARGS
}

// ---------------- layer norm over rows ----------------
__global__ __launch_bounds__(256) void ln_rows_kernel(
    const float* __restrict__ X, long long strideX,
    float* __restrict__ Y, long long strideY,
    const float* __restrict__ g, const float* __restrict__ b, int W)
{
    const long long row = blockIdx.x;
    const float* x = X + row * strideX;
    float* y = Y + row * strideY;
    __shared__ float red[256];
    float s = 0.0f;
    for (int i = threadIdx.x; i < W; i += 256) s += x[i];
    red[threadIdx.x] = s; __syncthreads();
    for (int o = 128; o > 0; o >>= 1) {
        if (threadIdx.x < o) red[threadIdx.x] += red[threadIdx.x + o];
        __syncthreads();
    }
    const float mean = red[0] / (float)W;
    __syncthreads();
    s = 0.0f;
    for (int i = threadIdx.x; i < W; i += 256) { float d = x[i] - mean; s += d * d; }
    red[threadIdx.x] = s; __syncthreads();
    for (int o = 128; o > 0; o >>= 1) {
        if (threadIdx.x < o) red[threadIdx.x] += red[threadIdx.x + o];
        __syncthreads();
    }
    const float rstd = rsqrtf(red[0] / (float)W + LN_EPS_);
    for (int i = threadIdx.x; i < W; i += 256)
        y[i] = (x[i] - mean) * rstd * g[i] + b[i];
}

// ---------------- patch tensor build (5-way shift concat) ----------------
__global__ __launch_bounds__(256) void patch_build_kernel(
    const float* __restrict__ img, float* __restrict__ T)
{
    const int bp = blockIdx.x;                 // b*NPATCH + patch
    const int b = bp / NPATCH_, p = bp % NPATCH_;
    const int gy = p / GRID_, gx = p % GRID_;
    for (int f = threadIdx.x; f < PATCH_DIM_; f += 256) {
        const int pix = f / 15, c15 = f % 15;
        const int py = pix / P_, px = pix % P_;
        const int gset = c15 / 3, ch = c15 % 3;
        int h = gy * P_ + py, w = gx * P_ + px;
        if      (gset == 1) w -= 1;
        else if (gset == 2) w += 1;
        else if (gset == 3) h -= 1;
        else if (gset == 4) h += 1;
        float v = 0.0f;
        if (h >= 0 && h < IMG_ && w >= 0 && w < IMG_)
            v = img[((long long)b * C_ + ch) * IMG_ * IMG_ + (long long)h * IMG_ + w];
        T[(long long)bp * PATCH_DIM_ + f] = v;
    }
}

// ---------------- x = [cls; patches] + pos ----------------
__global__ void assemble_x_kernel(const float* __restrict__ Temb,
                                  const float* __restrict__ cls_tok,
                                  const float* __restrict__ pos,
                                  float* __restrict__ x)
{
    const long long i = (long long)blockIdx.x * 256 + threadIdx.x;
    if (i >= (long long)B_ * SEQ_ * D_) return;
    const int d = (int)(i % D_);
    const long long bn = i / D_;
    const int n = (int)(bn % SEQ_);
    const long long b = bn / SEQ_;
    float v = (n == 0) ? cls_tok[d]
                       : Temb[((long long)b * NPATCH_ + (n - 1)) * D_ + d];
    x[i] = v + pos[(long long)n * D_ + d];
}

// ---------------- attention softmax with diag mask + temperature ----------------
__global__ __launch_bounds__(256) void attn_softmax_kernel(
    float* __restrict__ S, const float* __restrict__ temp)
{
    const long long row = blockIdx.x;          // (b*H + h)*SEQ + n
    const int n = (int)(row % SEQ_);
    const float scale = expf(temp[0]);
    float* s = S + row * SLD_;
    __shared__ float red[256];
    float mx = -3.402823466e38f;
    for (int m = threadIdx.x; m < SEQ_; m += 256) {
        const float v = (m == n) ? -3.402823466e38f : s[m] * scale;
        mx = fmaxf(mx, v);
    }
    red[threadIdx.x] = mx; __syncthreads();
    for (int o = 128; o > 0; o >>= 1) {
        if (threadIdx.x < o) red[threadIdx.x] = fmaxf(red[threadIdx.x], red[threadIdx.x + o]);
        __syncthreads();
    }
    mx = red[0]; __syncthreads();
    float sum = 0.0f;
    for (int m = threadIdx.x; m < SEQ_; m += 256) {
        const float v = (m == n) ? 0.0f : expf(s[m] * scale - mx);
        s[m] = v;
        sum += v;
    }
    red[threadIdx.x] = sum; __syncthreads();
    for (int o = 128; o > 0; o >>= 1) {
        if (threadIdx.x < o) red[threadIdx.x] += red[threadIdx.x + o];
        __syncthreads();
    }
    const float inv = 1.0f / red[0];
    for (int m = threadIdx.x; m < SEQ_; m += 256) s[m] *= inv;
}

// ---------------- ctx = [mean_n x, max_n x] ----------------
__global__ void ctx_pool_kernel(const float* __restrict__ x, float* __restrict__ ctx)
{
    const int i = blockIdx.x * 256 + threadIdx.x;
    if (i >= B_ * D_) return;
    const int b = i / D_, d = i % D_;
    const float* p = x + (long long)b * SEQ_ * D_ + d;
    float s = 0.0f, mx = -3.402823466e38f;
    for (int n = 0; n < SEQ_; ++n) {
        const float v = p[(long long)n * D_];
        s += v; mx = fmaxf(mx, v);
    }
    ctx[(long long)b * 2 * D_ + d]      = s / (float)SEQ_;
    ctx[(long long)b * 2 * D_ + D_ + d] = mx;
}

// ---------------- C[r, :] += v (row-broadcast add) ----------------
__global__ void add_rowvec_kernel(float* __restrict__ Cx, const float* __restrict__ v,
                                  long long rows, int cols)
{
    const long long i = (long long)blockIdx.x * 256 + threadIdx.x;
    if (i >= rows * (long long)cols) return;
    Cx[i] += v[i % cols];
}

// ---------------- top-k selection over dyn expert scores ----------------
__global__ __launch_bounds__(64) void topk_mask_kernel(
    const float* __restrict__ Abuf, float* __restrict__ mask)
{
    const int b = blockIdx.x;
    __shared__ float sc[E_DYN_];
    const int e = threadIdx.x;
    if (e < E_DYN_) {
        float acc = 0.0f;
        for (int n = 0; n < SEQ_; ++n)
            acc += Abuf[((long long)b * SEQ_ + n) * E_TOT_ + E_FIX_ + e];
        sc[e] = acc;
    }
    __syncthreads();
    if (threadIdx.x == 0) {
        bool used[E_DYN_];
        for (int j = 0; j < E_DYN_; ++j) used[j] = false;
        for (int i = 0; i < TOPK_; ++i) {
            int bi = 0; float bv = -3.402823466e38f;
            for (int j = 0; j < E_DYN_; ++j)
                if (!used[j] && sc[j] > bv) { bv = sc[j]; bi = j; }
            used[bi] = true;
        }
        for (int j = 0; j < E_FIX_; ++j) mask[b * E_TOT_ + j] = 1.0f;
        for (int j = 0; j < E_DYN_; ++j)
            mask[b * E_TOT_ + E_FIX_ + j] = used[j] ? 1.0f : 0.0f;
    }
}

// ---------------- softmax over tokens per (b,e) with mask-scaled logits ----------------
__global__ __launch_bounds__(256) void head_softmax_kernel(
    float* __restrict__ Abuf, const float* __restrict__ mask)
{
    const int be = blockIdx.x;
    const int b = be / E_TOT_, e = be % E_TOT_;
    const float mk = mask[b * E_TOT_ + e];
    float* base = Abuf + (long long)b * SEQ_ * E_TOT_ + e;
    __shared__ float red[256];
    float mx = -3.402823466e38f;
    for (int n = threadIdx.x; n < SEQ_; n += 256)
        mx = fmaxf(mx, base[(long long)n * E_TOT_] * mk);
    red[threadIdx.x] = mx; __syncthreads();
    for (int o = 128; o > 0; o >>= 1) {
        if (threadIdx.x < o) red[threadIdx.x] = fmaxf(red[threadIdx.x], red[threadIdx.x + o]);
        __syncthreads();
    }
    mx = red[0]; __syncthreads();
    float sum = 0.0f;
    for (int n = threadIdx.x; n < SEQ_; n += 256) {
        const float v = expf(base[(long long)n * E_TOT_] * mk - mx);
        base[(long long)n * E_TOT_] = v;
        sum += v;
    }
    red[threadIdx.x] = sum; __syncthreads();
    for (int o = 128; o > 0; o >>= 1) {
        if (threadIdx.x < o) red[threadIdx.x] += red[threadIdx.x + o];
        __syncthreads();
    }
    const float inv = 1.0f / red[0];
    for (int n = threadIdx.x; n < SEQ_; n += 256) base[(long long)n * E_TOT_] *= inv;
}

// =========================================================================
extern "C" void kernel_launch(void* const* d_in, const int* in_sizes, int n_in,
                              void* d_out, int out_size, void* d_ws, size_t ws_size,
                              hipStream_t stream)
{
    (void)in_sizes; (void)n_in; (void)out_size; (void)ws_size;

    const float* img        = (const float*)d_in[0];
    const float* patch_ln_g = (const float*)d_in[1];
    const float* patch_ln_b = (const float*)d_in[2];
    const float* patch_w    = (const float*)d_in[3];
    const float* patch_b    = (const float*)d_in[4];
    const float* pos_emb    = (const float*)d_in[5];
    const float* cls_tok    = (const float*)d_in[6];
    const float* ln1_g      = (const float*)d_in[7];
    const float* ln1_b      = (const float*)d_in[8];
    const float* qkv_w      = (const float*)d_in[9];
    const float* attn_out_w = (const float*)d_in[10];
    const float* attn_out_b = (const float*)d_in[11];
    const float* temperature= (const float*)d_in[12];
    const float* ln2_g      = (const float*)d_in[13];
    const float* ln2_b      = (const float*)d_in[14];
    const float* ff_w1      = (const float*)d_in[15];
    const float* ff_b1      = (const float*)d_in[16];
    const float* ff_w2      = (const float*)d_in[17];
    const float* ff_b2      = (const float*)d_in[18];
    const float* proto_base = (const float*)d_in[19];
    const float* ctx_w      = (const float*)d_in[20];
    const float* ctx_b      = (const float*)d_in[21];
    const float* pre_w      = (const float*)d_in[22];
    const float* pre_b      = (const float*)d_in[23];
    const float* edge_w     = (const float*)d_in[24];
    const float* edge_b     = (const float*)d_in[25];
    const float* node_w     = (const float*)d_in[26];
    const float* node_b     = (const float*)d_in[27];
    const float* head_ln_g  = (const float*)d_in[28];
    const float* head_ln_b  = (const float*)d_in[29];
    const float* head_w     = (const float*)d_in[30];
    const float* head_b     = (const float*)d_in[31];
    float* out = (float*)d_out;

    // ---- workspace layout (floats) ----
    const long long TOK    = (long long)B_ * SEQ_;                  // 6304
    const long long SZ_X   = TOK * D_;                              // 4,841,472
    const long long SZ_QKV = TOK * 3 * D_;                          // 14,524,416

    float* ws   = (float*)d_ws;
    float* x    = ws;                    // persistent activations
    float* hbuf = x    + SZ_X;           // LN output / attn-o / Temb
    float* qkv  = hbuf + SZ_X;           // qkv; head-phase overlay region
    float* big  = qkv  + SZ_QKV;         // patch T / attn scores / ff1 / T2

    // head-phase overlay inside qkv region
    float* Xp     = qkv;                                    // SZ_X
    float* protos = Xp + SZ_X;                              // B*48*768
    float* Abuf   = protos + (long long)B_ * E_TOT_ * D_;   // B*197*48
    float* He     = Abuf + (long long)B_ * SEQ_ * E_TOT_;   // B*48*768
    float* T1     = He   + (long long)B_ * E_TOT_ * D_;     // B*48*768
    float* maskb  = T1   + (long long)B_ * E_TOT_ * D_;     // B*48
    float* ctx    = maskb + (long long)B_ * E_TOT_;         // B*1536
    float* cls    = ctx  + (long long)B_ * 2 * D_;          // B*768

    // ===================== patch embedding =====================
    patch_build_kernel<<<B_ * NPATCH_, 256, 0, stream>>>(img, big);
    ln_rows_kernel<<<B_ * NPATCH_, 256, 0, stream>>>(
        big, PATCH_DIM_, big, PATCH_DIM_, patch_ln_g, patch_ln_b, PATCH_DIM_);
    launch_gemm(stream, big, patch_w, patch_b, nullptr, hbuf,
                B_ * NPATCH_, D_, PATCH_DIM_, PATCH_DIM_, D_, D_,
                0, 0, 1.0f, 0, 1, 1, 0, 0, 0, 0, 0, 0);
    assemble_x_kernel<<<(int)((SZ_X + 255) / 256), 256, 0, stream>>>(
        hbuf, cls_tok, pos_emb, x);

    // ===================== transformer layers =====================
    const long long headA  = (long long)SEQ_ * 3 * D_;   // per-batch stride in qkv
    const long long scoreH = (long long)SEQ_ * SLD_;     // per-head score tile
    for (int l = 0; l < DEPTH_; ++l) {
        // h = LN1(x)
        ln_rows_kernel<<<(int)TOK, 256, 0, stream>>>(
            x, D_, hbuf, D_, ln1_g + (long long)l * D_, ln1_b + (long long)l * D_, D_);
        // qkv = h @ qkv_w[l]
        launch_gemm(stream, hbuf, qkv_w + (long long)l * D_ * 3 * D_, nullptr, nullptr, qkv,
                    (int)TOK, 3 * D_, D_, D_, 3 * D_, 3 * D_,
                    0, 0, 1.0f, 0, 1, 1, 0, 0, 0, 0, 0, 0);
        // scores[b,h] = Q @ K^T  (batched over b*h), ld = SLD_
        launch_gemm(stream, qkv /*Q*/, qkv + D_ /*K*/, nullptr, nullptr, big,
                    SEQ_, SEQ_, DH_, 3 * D_, 3 * D_, SLD_,
                    0, 1, 1.0f, 0, B_ * HEADS_, HEADS_,
                    headA, DH_, headA, DH_,
                    (long long)HEADS_ * scoreH, scoreH);
        attn_softmax_kernel<<<B_ * HEADS_ * SEQ_, 256, 0, stream>>>(big, temperature + l);
        // o[b,:,h*64..] = attn @ V
        launch_gemm(stream, big, qkv + 2 * D_ /*V*/, nullptr, nullptr, hbuf,
                    SEQ_, DH_, SEQ_, SLD_, 3 * D_, D_,
                    0, 0, 1.0f, 0, B_ * HEADS_, HEADS_,
                    (long long)HEADS_ * scoreH, scoreH,
                    headA, DH_,
                    (long long)SEQ_ * D_, DH_);
        // x = x + o @ attn_out_w + attn_out_b
        launch_gemm(stream, hbuf, attn_out_w + (long long)l * D_ * D_,
                    attn_out_b + (long long)l * D_, x, x,
                    (int)TOK, D_, D_, D_, D_, D_,
                    0, 0, 1.0f, 0, 1, 1, 0, 0, 0, 0, 0, 0);
        // h = LN2(x)
        ln_rows_kernel<<<(int)TOK, 256, 0, stream>>>(
            x, D_, hbuf, D_, ln2_g + (long long)l * D_, ln2_b + (long long)l * D_, D_);
        // ff1 = gelu(h @ w1 + b1)
        launch_gemm(stream, hbuf, ff_w1 + (long long)l * D_ * MLP_,
                    ff_b1 + (long long)l * MLP_, nullptr, big,
                    (int)TOK, MLP_, D_, D_, MLP_, MLP_,
                    0, 0, 1.0f, 1, 1, 1, 0, 0, 0, 0, 0, 0);
        // x = x + ff1 @ w2 + b2
        launch_gemm(stream, big, ff_w2 + (long long)l * MLP_ * D_,
                    ff_b2 + (long long)l * D_, x, x,
                    (int)TOK, D_, MLP_, MLP_, D_, D_,
                    0, 0, 1.0f, 0, 1, 1, 0, 0, 0, 0, 0, 0);
    }

    // ===================== hypergraph head =====================
    ctx_pool_kernel<<<(B_ * D_ + 255) / 256, 256, 0, stream>>>(x, ctx);
    launch_gemm(stream, ctx, ctx_w, ctx_b, nullptr, protos,
                B_, E_TOT_ * D_, 2 * D_, 2 * D_, E_TOT_ * D_, E_TOT_ * D_,
                0, 0, 1.0f, 0, 1, 1, 0, 0, 0, 0, 0, 0);
    add_rowvec_kernel<<<(int)(((long long)B_ * E_TOT_ * D_ + 255) / 256), 256, 0, stream>>>(
        protos, proto_base, B_, E_TOT_ * D_);
    launch_gemm(stream, x, pre_w, pre_b, nullptr, Xp,
                (int)TOK, D_, D_, D_, D_, D_,
                0, 0, 1.0f, 0, 1, 1, 0, 0, 0, 0, 0, 0);
    {   // A[b,n,e] = (Xp[b,n,:] . protos[b,e,:]) / (EH*sqrt(EHD))
        const float ascale = 1.0f / ((float)EH_ * sqrtf((float)EHD_));
        launch_gemm(stream, Xp, protos, nullptr, nullptr, Abuf,
                    SEQ_, E_TOT_, D_, D_, D_, E_TOT_,
                    0, 1, ascale, 0, B_, 1,
                    (long long)SEQ_ * D_, 0,
                    (long long)E_TOT_ * D_, 0,
                    (long long)SEQ_ * E_TOT_, 0);
    }
    topk_mask_kernel<<<B_, 64, 0, stream>>>(Abuf, maskb);
    head_softmax_kernel<<<B_ * E_TOT_, 256, 0, stream>>>(Abuf, maskb);
    // T1[b,e,:] = A[b]^T @ x[b]
    launch_gemm(stream, Abuf, x, nullptr, nullptr, T1,
                E_TOT_, D_, SEQ_, E_TOT_, D_, D_,
                1, 0, 1.0f, 0, B_, 1,
                (long long)SEQ_ * E_TOT_, 0,
                (long long)SEQ_ * D_, 0,
                (long long)E_TOT_ * D_, 0);
    // He = gelu(T1 @ edge_w + edge_b)
    launch_gemm(stream, T1, edge_w, edge_b, nullptr, He,
                B_ * E_TOT_, D_, D_, D_, D_, D_,
                0, 0, 1.0f, 1, 1, 1, 0, 0, 0, 0, 0, 0);
    // T2[b,n,:] = A[b] @ He[b]  -> big
    launch_gemm(stream, Abuf, He, nullptr, nullptr, big,
                SEQ_, D_, E_TOT_, E_TOT_, D_, D_,
                0, 0, 1.0f, 0, B_, 1,
                (long long)SEQ_ * E_TOT_, 0,
                (long long)E_TOT_ * D_, 0,
                (long long)SEQ_ * D_, 0);
    // x = gelu(T2 @ node_w + node_b) + x
    launch_gemm(stream, big, node_w, node_b, x, x,
                (int)TOK, D_, D_, D_, D_, D_,
                0, 0, 1.0f, 1, 1, 1, 0, 0, 0, 0, 0, 0);
    // cls = LN(x[:,0,:])
    ln_rows_kernel<<<B_, 256, 0, stream>>>(
        x, (long long)SEQ_ * D_, cls, D_, head_ln_g, head_ln_b, D_);
    // out = cls @ head_w + head_b
    launch_gemm(stream, cls, head_w, head_b, nullptr, out,
                B_, NCLS_, D_, D_, NCLS_, NCLS_,
                0, 0, 1.0f, 0, 1, 1, 0, 0, 0, 0, 0, 0);
}
